// SimpleStructLearner_70377334113124
// MI455X (gfx1250) — compile-verified, matrix operand order checked by reading, not run on hardware
//
#include <hip/hip_runtime.h>
#include <hip/hip_fp16.h>
#include <math.h>

typedef __attribute__((ext_vector_type(16))) _Float16 v16h;
typedef __attribute__((ext_vector_type(8)))  _Float16 v8h;
typedef __attribute__((ext_vector_type(8)))  float    v8f;
typedef __attribute__((ext_vector_type(2)))  _Float16 h2;

#define FEAT 64
#define HID  128
#define KDIM 128   // 2*FEAT == HID == 128

// ---------------------------------------------------------------------------
// Prologue: convert node features f32 -> f16 once (25.6 MB -> 12.8 MB, stays
// L2-resident; halves gather traffic and removes all cvt VALU from hot loop).
// ---------------------------------------------------------------------------
__global__ __launch_bounds__(256) void convert_x_kernel(
    const float* __restrict__ x, _Float16* __restrict__ xh, int n)
{
  int stride = gridDim.x * blockDim.x;
  for (int i = blockIdx.x * blockDim.x + threadIdx.x; i < n; i += stride)
    xh[i] = (_Float16)x[i];
}

// ---------------------------------------------------------------------------
// Main kernel: per-wave 16-edge tile.
//   h_edge[16 x 128] = concat(x[src], x[dst])      (A fragments)
//   h = relu(h_edge @ W1 + b1)                     (32x v_wmma_f32_16x16x32_f16,
//                                                   B fragments hoisted into
//                                                   extended VGPRs by compiler)
//   logit = h @ W2 + b2 ; out = sigmoid(logit)     (VALU + shfl_xor reduction)
// b1 is folded into the accumulator init (bias depends only on column n).
// ---------------------------------------------------------------------------
template <bool USE_F16_X>
__global__ __launch_bounds__(256) void edge_mlp_kernel(
    const float*    __restrict__ x,    // f32 node table (fallback path)
    const _Float16* __restrict__ xh,   // f16 node table (fast path)
    const int*      __restrict__ src,
    const int*      __restrict__ dst,
    const float*    __restrict__ W1,   // [128 (K) x 128 (N)] row-major
    const float*    __restrict__ b1,   // [128]
    const float*    __restrict__ W2,   // [128]
    const float*    __restrict__ b2,   // [1]
    float*          __restrict__ out,  // [E]
    int E, int totalTiles)
{
  __shared__ _Float16 w1t[KDIM * HID];   // transposed: w1t[n*128 + k]
  __shared__ float    b1s[HID];
  __shared__ float    w2s[HID];

  for (int i = threadIdx.x; i < KDIM * HID; i += blockDim.x) {
    int k = i >> 7, n = i & 127;
    w1t[n * KDIM + k] = (_Float16)W1[i];
  }
  for (int i = threadIdx.x; i < HID; i += blockDim.x) {
    b1s[i] = b1[i];
    w2s[i] = W2[i];
  }
  __syncthreads();

  const int lane  = threadIdx.x & 31;
  const int wave  = threadIdx.x >> 5;
  const int mrow  = lane & 15;   // A-matrix: lane -> row M
  const int khalf = lane >> 4;   // A/B: second lane-half covers K+8
  const int ncol  = lane & 15;   // B/C/D: lane -> column N (within tile)
  const float bias2 = b2[0];

  // Per-lane loop-invariant layer-1 bias / layer-2 weight for each N-tile.
  float biasv[8], w2v[8];
#pragma unroll
  for (int t = 0; t < 8; ++t) {
    biasv[t] = b1s[t * 16 + ncol];
    w2v[t]   = w2s[t * 16 + ncol];
  }

  for (int tileIdx = blockIdx.x * 8 + wave; tileIdx < totalTiles;
       tileIdx += gridDim.x * 8) {
    const int e0 = tileIdx * 16;
    int e = e0 + mrow; if (e >= E) e = E - 1;          // clamp (keeps EXEC full)
    const int si = src[e];
    const int di = dst[e];

    // ---- Build A fragments: 4 chunks of K=32 over the 128 concat features.
    // Chunks 0..1 = src features (cols 0..63), chunks 2..3 = dst features.
    v16h a[4];
    if (USE_F16_X) {
      const _Float16* srow = xh + (long long)si * FEAT;
      const _Float16* drow = xh + (long long)di * FEAT;
#pragma unroll
      for (int chunk = 0; chunk < 4; ++chunk) {
        const _Float16* rowp = (chunk < 2) ? srow : drow;
        const int coff = (chunk & 1) * 32 + khalf * 8;
        v8h lo = *(const v8h*)(rowp + coff);        // global_load_b128
        v8h hi = *(const v8h*)(rowp + coff + 16);   // global_load_b128
#pragma unroll
        for (int i = 0; i < 8; ++i) {
          a[chunk][i]     = lo[i];
          a[chunk][8 + i] = hi[i];
        }
      }
    } else {
      const float* srow = x + (long long)si * FEAT;
      const float* drow = x + (long long)di * FEAT;
#pragma unroll
      for (int chunk = 0; chunk < 4; ++chunk) {
        const float* rowp = (chunk < 2) ? srow : drow;
        const int coff = (chunk & 1) * 32 + khalf * 8;
#pragma unroll
        for (int v = 0; v < 8; ++v) {
          int c = coff + (v >> 2) * 16 + (v & 3) * 2;
          float2 f = *(const float2*)(rowp + c);
          a[chunk][2 * v]     = (_Float16)f.x;
          a[chunk][2 * v + 1] = (_Float16)f.y;
        }
      }
    }

    float partial[8];
#pragma unroll
    for (int r = 0; r < 8; ++r) partial[r] = 0.f;

    // ---- Layer 1: 8 N-tiles of 16 cols, K=128 in 4 WMMA steps each.
#pragma unroll
    for (int t = 0; t < 8; ++t) {
      const int nglob = t * 16 + ncol;
      v8f acc;
#pragma unroll
      for (int r = 0; r < 8; ++r) acc[r] = biasv[t];   // bias folded into C
#pragma unroll
      for (int chunk = 0; chunk < 4; ++chunk) {
        v16h b;
        const int kb = chunk * 32 + khalf * 8;
#pragma unroll
        for (int v = 0; v < 8; ++v) {
          int K = kb + (v >> 2) * 16 + (v & 3) * 2;
          h2 p = *(const h2*)&w1t[nglob * KDIM + K];   // hoisted by compiler
          b[2 * v]     = p[0];
          b[2 * v + 1] = p[1];
        }
        acc = __builtin_amdgcn_wmma_f32_16x16x32_f16(
            /*neg_a=*/false, a[chunk], /*neg_b=*/false, b,
            /*c_mod=*/(short)0, acc, /*reuse_a=*/false, /*reuse_b=*/false);
      }
      // ReLU + layer-2 partial dot (same column n for all 8 rows).
#pragma unroll
      for (int r = 0; r < 8; ++r) {
        float hv = acc[r] > 0.f ? acc[r] : 0.f;
        partial[r] += hv * w2v[t];
      }
    }

    // ---- Reduce layer-2 partials over the 16 columns (lanes within a half).
#pragma unroll
    for (int off = 1; off < 16; off <<= 1) {
#pragma unroll
      for (int r = 0; r < 8; ++r)
        partial[r] += __shfl_xor(partial[r], off, 32);
    }

    // C/D layout: VGPR r of lane-half kh holds row m = r + 8*kh.
    // lanes 0..7 emit edges e0+0..7, lanes 16..23 emit e0+8..15.
    float myv = 0.f;
#pragma unroll
    for (int r = 0; r < 8; ++r)
      if ((lane & 7) == r) myv = partial[r];
    const int eo = e0 + khalf * 8 + (lane & 7);
    if ((lane & 15) < 8 && eo < E) {
      float z = myv + bias2;
      out[eo] = 1.0f / (1.0f + __expf(-z));
    }
  }
}

// ---------------------------------------------------------------------------
// Deterministic two-stage reduction for sparsity_loss = 1e-3 * mean(|w|).
// ---------------------------------------------------------------------------
__global__ __launch_bounds__(256) void reduce_partial_kernel(
    const float* __restrict__ w, float* __restrict__ partials, int E)
{
  __shared__ float red[256];
  const int chunk = (E + gridDim.x - 1) / gridDim.x;
  const int beg = blockIdx.x * chunk;
  int end = beg + chunk; if (end > E) end = E;
  float s = 0.f;
  for (int i = beg + threadIdx.x; i < end; i += 256) s += fabsf(w[i]);
  red[threadIdx.x] = s;
  __syncthreads();
  for (int o = 128; o > 0; o >>= 1) {
    if (threadIdx.x < o) red[threadIdx.x] += red[threadIdx.x + o];
    __syncthreads();
  }
  if (threadIdx.x == 0) partials[blockIdx.x] = red[0];
}

__global__ __launch_bounds__(256) void reduce_final_kernel(
    const float* __restrict__ partials, int nPart,
    float* __restrict__ loss, int E)
{
  __shared__ float red[256];
  float s = 0.f;
  for (int i = threadIdx.x; i < nPart; i += 256) s += partials[i];
  red[threadIdx.x] = s;
  __syncthreads();
  for (int o = 128; o > 0; o >>= 1) {
    if (threadIdx.x < o) red[threadIdx.x] += red[threadIdx.x + o];
    __syncthreads();
  }
  if (threadIdx.x == 0) *loss = 1.0e-3f * red[0] / (float)E;
}

extern "C" void kernel_launch(void* const* d_in, const int* in_sizes, int n_in,
                              void* d_out, int out_size, void* d_ws, size_t ws_size,
                              hipStream_t stream) {
  const float* x  = (const float*)d_in[0];
  const int*   ei = (const int*)  d_in[1];   // [2, E] row-major
  const float* W1 = (const float*)d_in[2];
  const float* b1 = (const float*)d_in[3];
  const float* W2 = (const float*)d_in[4];
  const float* b2 = (const float*)d_in[5];
  float* out = (float*)d_out;                // [E] weights, then [1] loss

  const int E = in_sizes[1] / 2;
  const int nX = in_sizes[0];                // N_NODES * FEAT
  const int* src = ei;
  const int* dst = ei + E;

  const int totalTiles = (E + 15) / 16;      // 16 edges per wave-tile
  int blocks = (totalTiles + 7) / 8;         // 8 waves per 256-thread block
  if (blocks > 2048) blocks = 2048;          // grid-stride amortizes LDS fill

  // Workspace layout: [0,1KB) reduction partials, [2KB, 2KB+nX*2) f16 x table.
  const size_t need = 2048 + (size_t)nX * sizeof(_Float16);
  _Float16* xh = (_Float16*)((char*)d_ws + 2048);
  float* partials = (float*)d_ws;

  if (ws_size >= need) {
    convert_x_kernel<<<256, 256, 0, stream>>>(x, xh, nX);
    edge_mlp_kernel<true><<<blocks, 256, 0, stream>>>(
        x, xh, src, dst, W1, b1, W2, b2, out, E, totalTiles);
  } else {
    edge_mlp_kernel<false><<<blocks, 256, 0, stream>>>(
        x, (const _Float16*)nullptr, src, dst, W1, b1, W2, b2, out, E,
        totalTiles);
  }

  const int NPART = 256;
  reduce_partial_kernel<<<NPART, 256, 0, stream>>>(out, partials, E);
  reduce_final_kernel<<<1, 256, 0, stream>>>(partials, NPART, out + E, E);
}
